// Rocket_60266981097926
// MI455X (gfx1250) — matrix operand order repaired and checked by reading, not compiled, not run on hardware
//
#include <hip/hip_runtime.h>
#include <hip/hip_bf16.h>
#include <stdint.h>

#define NK    100
#define NF    64
#define SEQ   1024
#define NCLS  10
#define BATCH 512
#define BLOCK 512          // 16 waves (wave32)
#define NWAVE (BLOCK / 32)
#define SY    (SEQ + 4)    // Yt row stride (floats): 1028 -> bank-conflict-free b128 stores

typedef __attribute__((ext_vector_type(16))) __bf16 v16bf;
typedef __attribute__((ext_vector_type(8)))  float  v8f;

struct Ptrs100 { const void* p[NK]; };
struct Ks100   { int ks[NK]; };

__device__ __forceinline__ unsigned short f2bf(float f) {
  union { float f; unsigned u; } x; x.f = f;
  unsigned u = x.u;
  return (unsigned short)((u + 0x7FFFu + ((u >> 16) & 1u)) >> 16);
}

// ---------------------------------------------------------------------------
// Pack ragged conv weights (1,64,ks) into bf16 Wpad[i][k=0..15][c=0..63]
// (taps zero-padded to 16 => dense N for WMMA B fragments), and gather
// dilation/padding scalars into contiguous int arrays.
// ---------------------------------------------------------------------------
__global__ void rocket_prep(Ptrs100 wptrs, Ks100 K, unsigned short* __restrict__ Wpad,
                            Ptrs100 dptrs, Ptrs100 pptrs,
                            int* __restrict__ dil, int* __restrict__ pad) {
  const int i = blockIdx.x;
  const int c = threadIdx.x;                   // 0..63
  const float* w = (const float*)wptrs.p[i];   // layout [c][ks]
  const int ks = K.ks[i];
#pragma unroll
  for (int k = 0; k < 16; ++k) {
    float v = (k < ks) ? w[c * ks + k] : 0.0f;
    Wpad[(i * 16 + k) * NF + c] = f2bf(v);
  }
  if (c == 0) {
    dil[i] = *(const int*)dptrs.p[i];
    pad[i] = *(const int*)pptrs.p[i];
  }
}

// ---------------------------------------------------------------------------
// One block per batch row b. LDS: xs = x_b transposed as packed bf16 pairs
// (128KB), two transposed GEMM slabs Yt[k][l]. Two conv kernels per pass
// share A fragments: 4 interleaved (hazard-free) v_wmma_f32_16x16x32_bf16
// per tile, D written as ds_store_b128 pairs. Then shifted gather-sum ->
// max / PPV features, then 200->10 FC.
// ---------------------------------------------------------------------------
__global__ __launch_bounds__(BLOCK) void rocket_main(
    const float* __restrict__ x,              // (512,64,1024)
    const unsigned short* __restrict__ Wpad,  // (100,16,64) bf16
    const int* __restrict__ dil, const int* __restrict__ pad,
    const float* __restrict__ bias,           // (100)
    const float* __restrict__ fcw,            // (10,200)
    const float* __restrict__ fcb,            // (10)
    float* __restrict__ out,                  // (512,10)
    Ks100 K) {
  extern __shared__ unsigned smem[];
  unsigned* xs   = smem;                        // SEQ*32 dwords (packed bf16 c-pairs)
  float*    Yt0  = (float*)(smem + SEQ * 32);   // 16 x SY floats  (kernel i0)
  float*    Yt1  = Yt0 + 16 * SY;               // 16 x SY floats  (kernel i1)
  float*    red  = Yt1 + 16 * SY;               // 4*NWAVE floats
  float*    feat = red + 4 * NWAVE;             // 200 floats

  const int tid  = threadIdx.x;
  const int b    = blockIdx.x;
  const int lane = tid & 31;
  const int wv   = tid >> 5;
  const int g    = lane >> 4;                   // lane half (wave32 WMMA layout)
  const int hl   = lane & 15;
  const int wvu  = __builtin_amdgcn_readfirstlane(wv);  // wave-uniform -> scalar loop

  // ---- Phase A: stage x[b] into LDS, transposed + converted to bf16 -------
  const float* xb = x + (size_t)b * NF * SEQ;
  for (int idx = tid; idx < SEQ * 32; idx += BLOCK) {
    const int l  = idx >> 5;
    const int cp = idx & 31;
    const float a0 = xb[(2 * cp)     * SEQ + l];
    const float a1 = xb[(2 * cp + 1) * SEQ + l];
    xs[l * 32 + cp] = (unsigned)f2bf(a0) | ((unsigned)f2bf(a1) << 16);
  }
  __syncthreads();

  for (int j = 0; j < NK / 2; ++j) {
    const int i0 = 2 * j;
    const int i1 = 2 * j + 1;

    // ---- B fragments for both kernels: chunk0 = c 0..31, chunk1 = c 32..63
    // B lanes as loaded here (contiguous k, same as A): N = lane%16, K(half q) = q + 16*g.
    union BF { v16bf v; uint4 q[2]; } b00, b01, b10, b11;
    const unsigned short* w0 = Wpad + (size_t)(i0 * 16 + hl) * NF;
    const unsigned short* w1 = Wpad + (size_t)(i1 * 16 + hl) * NF;
    b00.q[0] = *(const uint4*)(w0 + 16 * g);
    b00.q[1] = *(const uint4*)(w0 + 16 * g + 8);
    b01.q[0] = *(const uint4*)(w0 + 32 + 16 * g);
    b01.q[1] = *(const uint4*)(w0 + 32 + 16 * g + 8);
    b10.q[0] = *(const uint4*)(w1 + 16 * g);
    b10.q[1] = *(const uint4*)(w1 + 16 * g + 8);
    b11.q[0] = *(const uint4*)(w1 + 32 + 16 * g);
    b11.q[1] = *(const uint4*)(w1 + 32 + 16 * g + 8);

    // ---- Phase B: Yt[k,l] = sum_c xs[l,c]*W[c,k] for both kernels ---------
    for (int tt = wvu; tt < SEQ / 16; tt += NWAVE) {
      const int l0 = tt * 16;
      // A layout (16x32 bf16): M = lane%16; per lane two contiguous 16B
      // chunks per 32-channel slice.
      const unsigned* row = xs + (l0 + hl) * 32;
      union AF { v16bf v; uint4 q[2]; } a0, a1;
      a0.q[0] = *(const uint4*)(row + 8 * g);            // c 0..31 slice
      a0.q[1] = *(const uint4*)(row + 8 * g + 4);
      a1.q[0] = *(const uint4*)(row + 16 + 8 * g);       // c 32..63 slice
      a1.q[1] = *(const uint4*)(row + 20 + 8 * g);

      v8f c0 = {}, c1 = {};
      // interleave the two kernels so adjacent WMMAs are independent
      c0 = __builtin_amdgcn_wmma_f32_16x16x32_bf16(false, a0.v, false, b00.v,
                                                   (short)0, c0, false, false);
      c1 = __builtin_amdgcn_wmma_f32_16x16x32_bf16(false, a0.v, false, b10.v,
                                                   (short)0, c1, false, false);
      c0 = __builtin_amdgcn_wmma_f32_16x16x32_bf16(false, a1.v, false, b01.v,
                                                   (short)0, c0, false, false);
      c1 = __builtin_amdgcn_wmma_f32_16x16x32_bf16(false, a1.v, false, b11.v,
                                                   (short)0, c1, false, false);

      // D layout: VGPR r -> (M = r + 8*g, N = lane%16): 8 consecutive rows
      // at fixed k => transposed store is two contiguous float4's per lane.
      union DF { v8f v; float4 f[2]; } d0, d1;
      d0.v = c0; d1.v = c1;
      float* y0 = Yt0 + hl * SY + l0 + 8 * g;
      float* y1 = Yt1 + hl * SY + l0 + 8 * g;
      *(float4*)(y0)     = d0.f[0];
      *(float4*)(y0 + 4) = d0.f[1];
      *(float4*)(y1)     = d1.f[0];
      *(float4*)(y1 + 4) = d1.f[1];
    }
    __syncthreads();

    // ---- Phase C: out[l] = bias + sum_k Yt[k, l + k*d - p]; max & PPV ----
#pragma unroll
    for (int m = 0; m < 2; ++m) {
      const float* Ym = m ? Yt1 : Yt0;
      const int i  = i0 + m;
      const int d  = dil[i];
      const int p  = pad[i];
      const int ks = K.ks[i];
      const int Lo = SEQ + 2 * p - d * (ks - 1);
      const float bi = bias[i];
      float lmax = -3.402823e38f;
      float cnt  = 0.0f;
      for (int l = tid; l < Lo; l += BLOCK) {
        float acc = bi;
        const int base = l - p;
        for (int k = 0; k < ks; ++k) {
          const int lp = base + k * d;
          if ((unsigned)lp < (unsigned)SEQ) acc += Ym[k * SY + lp];
        }
        lmax = fmaxf(lmax, acc);
        cnt += (acc > 0.0f) ? 1.0f : 0.0f;
      }
#pragma unroll
      for (int msk = 16; msk > 0; msk >>= 1) {
        lmax = fmaxf(lmax, __shfl_xor(lmax, msk, 32));
        cnt += __shfl_xor(cnt, msk, 32);
      }
      if (lane == 0) {
        red[m * 2 * NWAVE + wv]         = lmax;
        red[m * 2 * NWAVE + NWAVE + wv] = cnt;
      }
    }
    __syncthreads();
    if (tid == 0) {
#pragma unroll
      for (int m = 0; m < 2; ++m) {
        float gm = red[m * 2 * NWAVE];
        float gc = red[m * 2 * NWAVE + NWAVE];
        for (int w2 = 1; w2 < NWAVE; ++w2) {
          gm = fmaxf(gm, red[m * 2 * NWAVE + w2]);
          gc += red[m * 2 * NWAVE + NWAVE + w2];
        }
        const int i  = i0 + m;
        const int Lo = SEQ + 2 * pad[i] - dil[i] * (K.ks[i] - 1);
        feat[2 * i]     = gm;
        feat[2 * i + 1] = gc / (float)Lo;
      }
    }
    __syncthreads();   // Yt + feat consistent before next pair reuses slabs
  }

  // ---- Phase D: 200 -> 10 linear layer ------------------------------------
  if (tid < NCLS) {
    float s = fcb[tid];
    const float* wr = fcw + tid * (2 * NK);
    for (int jj = 0; jj < 2 * NK; ++jj) s += feat[jj] * wr[jj];
    out[b * NCLS + tid] = s;
  }
}

// ---------------------------------------------------------------------------
extern "C" void kernel_launch(void* const* d_in, const int* in_sizes, int n_in,
                              void* d_out, int out_size, void* d_ws, size_t ws_size,
                              hipStream_t stream) {
  (void)out_size; (void)ws_size;
  Ptrs100 wptrs, dptrs, pptrs;
  Ks100 K;

  const float* x = (const float*)d_in[0];
  for (int i = 0; i < NK; ++i) {
    wptrs.p[i] = d_in[1 + i];               // conv_weights list -> 100 inputs
    K.ks[i]    = in_sizes[1 + i] / NF;      // (1,64,ks) => ks from flat size
  }
  const float* bias = (const float*)d_in[101];
  const float* fcw  = (const float*)d_in[102];
  const float* fcb  = (const float*)d_in[103];

  if (n_in >= 304) {
    // dilations / paddings flattened as 100 scalar inputs each
    for (int i = 0; i < NK; ++i) {
      dptrs.p[i] = d_in[104 + i];
      pptrs.p[i] = d_in[204 + i];
    }
  } else if (n_in >= 106) {
    // dilations / paddings as single length-100 int arrays
    const int* dd = (const int*)d_in[104];
    const int* pp = (const int*)d_in[105];
    for (int i = 0; i < NK; ++i) { dptrs.p[i] = dd + i; pptrs.p[i] = pp + i; }
  } else {
    // fallback: last two inputs are the int arrays
    const int* dd = (const int*)d_in[n_in - 2];
    const int* pp = (const int*)d_in[n_in - 1];
    for (int i = 0; i < NK; ++i) { dptrs.p[i] = dd + i; pptrs.p[i] = pp + i; }
  }

  // Workspace layout: Wpad bf16 [100][16][64], then dil[100], pad[100]
  unsigned short* Wpad = (unsigned short*)d_ws;
  int* dil = (int*)((char*)d_ws + (size_t)NK * 16 * NF * 2);
  int* pad = dil + NK;

  rocket_prep<<<NK, NF, 0, stream>>>(wptrs, K, Wpad, dptrs, pptrs, dil, pad);

  const size_t smem = ((size_t)SEQ * 32            // xs: 128 KB packed bf16
                     + (size_t)2 * 16 * SY         // Yt0/Yt1: 2 x ~65.8 KB f32
                     + 4 * NWAVE                   // reduction scratch
                     + 2 * NK) * 4;                // features
  rocket_main<<<BATCH, BLOCK, smem, stream>>>(x, Wpad, dil, pad, bias, fcw, fcb,
                                              (float*)d_out, K);
}
